// Net_42769284333949
// MI455X (gfx1250) — matrix operand order repaired and checked by reading, not compiled, not run on hardware
//
#include <hip/hip_runtime.h>
#include <hip/hip_bf16.h>
#include <stdint.h>

typedef __attribute__((ext_vector_type(16))) _Float16 v16h;
typedef __attribute__((ext_vector_type(8)))  float    v8f;
typedef __attribute__((ext_vector_type(4)))  unsigned su4;
typedef __attribute__((ext_vector_type(8)))  unsigned su8;

#define NN  400000
#define EE  1200000
#define GG  20000
#define CC  10000
#define DIN 30
#define DH  64
#define DL  128

static __device__ __forceinline__ float relu_f(float v) { return v > 0.f ? v : 0.f; }

// ---------------------------------------------------------------------------
// Weight packing: W is Kact x Ncols row-major f32. Packed B-fragment layout
// per wave32 WMMA 16x16x32 f16: fragment (nt,kt): lane j -> column nt*16+(j&15),
// K rows kt*32 + (j>>4)*16 + {0..15}, stored as 16 f16 per lane (8 dword pairs).
// ---------------------------------------------------------------------------
__global__ void pack_w_kernel(const float* __restrict__ W, int Kact, int Ncols, int KT,
                              _Float16* __restrict__ dst) {
    int NT = Ncols >> 4;
    int total = NT * KT * 32 * 8;
    int t = blockIdx.x * blockDim.x + threadIdx.x;
    if (t >= total) return;
    int r    = t & 7;
    int id   = t >> 3;
    int lane = id & 31;
    int ft   = id >> 5;
    int kt   = ft % KT;
    int nt   = ft / KT;
    int col  = nt * 16 + (lane & 15);
    int k0   = kt * 32 + (lane >> 4) * 16 + 2 * r;
    float v0 = (k0     < Kact) ? W[(size_t)k0 * Ncols + col]       : 0.f;
    float v1 = (k0 + 1 < Kact) ? W[(size_t)(k0 + 1) * Ncols + col] : 0.f;
    dst[(size_t)t * 2]     = (_Float16)v0;
    dst[(size_t)t * 2 + 1] = (_Float16)v1;
}

// ---------------------------------------------------------------------------
// Utility kernels
// ---------------------------------------------------------------------------
__global__ void zero_f32_kernel(float* __restrict__ p, int n) {
    int t = blockIdx.x * blockDim.x + threadIdx.x;
    for (int i = t; i < n; i += gridDim.x * blockDim.x) p[i] = 0.f;
}

__global__ void copy_f32v4_kernel(const float* __restrict__ src, float* __restrict__ dst, int n4) {
    int t = blockIdx.x * blockDim.x + threadIdx.x;
    if (t >= n4) return;
    ((float4*)dst)[t] = ((const float4*)src)[t];
}

// x (N x 30) -> out (N x 32) zero padded
__global__ void padcopy_x_kernel(const float* __restrict__ x, float* __restrict__ out, int n) {
    int t = blockIdx.x * blockDim.x + threadIdx.x;
    if (t >= n) return;
    int i = t >> 5, c = t & 31;
    out[t] = (c < DIN) ? x[(size_t)i * DIN + c] : 0.f;
}

// aggr[dst] += in[src] over edges, 4 channels per thread
__global__ void edge_scatter_kernel(const int* __restrict__ ei, const float* __restrict__ in,
                                    int inStride, int D, float* __restrict__ out, int outStride) {
    int groups = (D + 3) >> 2;
    long long t = (long long)blockIdx.x * blockDim.x + threadIdx.x;
    if (t >= (long long)EE * groups) return;
    int e = (int)(t / groups);
    int g = (int)(t % groups);
    int src = ei[e];
    int dst = ei[EE + e];
    int c0 = g * 4;
    const float* ip = in + (size_t)src * inStride + c0;
    float*       op = out + (size_t)dst * outStride + c0;
#pragma unroll
    for (int i = 0; i < 4; ++i)
        if (c0 + i < D) atomicAdd(op + i, ip[i]);
}

// segment_sum: out[idx[r]] += in[r], D multiple of 4
__global__ void segment_pool_kernel(const int* __restrict__ idx, int rows,
                                    const float* __restrict__ in, int D,
                                    float* __restrict__ out) {
    int groups = D >> 2;
    long long t = (long long)blockIdx.x * blockDim.x + threadIdx.x;
    if (t >= (long long)rows * groups) return;
    int r = (int)(t / groups);
    int g = (int)(t % groups);
    int seg = idx[r];
    int c0 = g * 4;
    float4 v = *(const float4*)(in + (size_t)r * D + c0);
    float* op = out + (size_t)seg * D + c0;
    atomicAdd(op + 0, v.x); atomicAdd(op + 1, v.y);
    atomicAdd(op + 2, v.z); atomicAdd(op + 3, v.w);
}

// fold BN stats into per-channel scale/shift
__global__ void bn_finalize_kernel(const float* __restrict__ stats, const float* __restrict__ g,
                                   const float* __restrict__ beta, float* __restrict__ ss,
                                   float invN) {
    int c = threadIdx.x;
    if (c >= DH) return;
    float mean = stats[c] * invN;
    float var  = stats[DH + c] * invN - mean * mean;
    float sc   = g[c] * rsqrtf(var + 1e-5f);
    ss[c]      = sc;
    ss[DH + c] = beta[c] - mean * sc;
}

// out[c] = sigmoid(dot(hc[c,:128], w) + b) : one wave per combination
__global__ void final_dot_kernel(const float* __restrict__ hc, const float* __restrict__ w,
                                 const float* __restrict__ b, float* __restrict__ out) {
    int wid  = (blockIdx.x * blockDim.x + threadIdx.x) >> 5;
    int lane = threadIdx.x & 31;
    if (wid >= CC) return;
    const float* row = hc + (size_t)wid * DL;
    float s = 0.f;
#pragma unroll
    for (int i = 0; i < 4; ++i) s += row[lane + 32 * i] * w[lane + 32 * i];
#pragma unroll
    for (int o = 16; o > 0; o >>= 1) s += __shfl_xor(s, o, 32);
    if (lane == 0) out[wid] = 1.f / (1.f + expf(-(s + b[0])));
}

// ---------------------------------------------------------------------------
// A-fragment loader: 16x32 f16 WMMA layout from f32 row-major, with optional
// fused per-K-channel affine (folded BatchNorm) + ReLU.
// lanes 0-15: K {0..15} in elems 0-15
// lanes 16-31: K {16..31} in elems 0-15
// ---------------------------------------------------------------------------
template <bool IN_AFFINE>
static __device__ __forceinline__ v16h load_afrag(const float* __restrict__ arow, int ktbase,
                                                  int sel, const float* __restrict__ ss) {
    v16h a;
#pragma unroll
    for (int half = 0; half < 2; ++half) {
        int kb = ktbase + sel * 16 + half * 8;
        float4 q0 = *(const float4*)(arow + kb);
        float4 q1 = *(const float4*)(arow + kb + 4);
        float vals[8] = {q0.x, q0.y, q0.z, q0.w, q1.x, q1.y, q1.z, q1.w};
        if (IN_AFFINE) {
            float4 s0 = *(const float4*)(ss + kb);
            float4 s1 = *(const float4*)(ss + kb + 4);
            float4 f0 = *(const float4*)(ss + DH + kb);
            float4 f1 = *(const float4*)(ss + DH + kb + 4);
            float sv[8] = {s0.x, s0.y, s0.z, s0.w, s1.x, s1.y, s1.z, s1.w};
            float fv[8] = {f0.x, f0.y, f0.z, f0.w, f1.x, f1.y, f1.z, f1.w};
#pragma unroll
            for (int i = 0; i < 8; ++i) vals[i] = relu_f(fmaf(vals[i], sv[i], fv[i]));
        }
#pragma unroll
        for (int i = 0; i < 8; ++i) a[half * 8 + i] = (_Float16)vals[i];
    }
    return a;
}

// ---------------------------------------------------------------------------
// WMMA GEMM: C[M x NCOLS] = op_in(A[M x KT*32]) @ B + bias.
// Packed weights are staged global->LDS once per workgroup with the Tensor
// Data Mover (tensor_load_to_lds, TENSORcnt), then each wave reads its
// B fragments from LDS. One wave per 16-row tile.
//   DO_STATS: store h = c+bias (no relu), accumulate per-channel sum/sumsq.
//   else:     store relu(c+bias).
// ---------------------------------------------------------------------------
template <int KT, int NCOLS, bool IN_AFFINE, bool DO_STATS>
__global__ void gemm_wmma_kernel(const float* __restrict__ A, int Mtiles,
                                 const _Float16* __restrict__ Bp,
                                 const float* __restrict__ bias, const float* __restrict__ ss,
                                 float* __restrict__ Co, float* __restrict__ stats) {
    constexpr int LDA   = KT * 32;
    constexpr int NT    = NCOLS / 16;
    constexpr int NHALF = NT * KT * 32 * 16;   // _Float16 elements in packed weights
    __shared__ __align__(32) _Float16 bsh[NHALF];

    // ---- TDM stage: packed weights -> LDS (wave 0 issues; TDM ignores EXEC) ----
    if (threadIdx.x < 32) {
        unsigned lds_off = (unsigned)(uintptr_t)(void*)bsh;   // addr[31:0] = LDS offset
        unsigned long long ga = (unsigned long long)(uintptr_t)Bp;
        const unsigned T = (NHALF * 2) / 8;                   // tile length in 8-byte units
        su4 g0; su8 g1;
        // D# group 0: count=1, lds_addr, global_addr[56:0], type=2
        g0[0] = 1u;
        g0[1] = lds_off;
        g0[2] = (unsigned)(ga & 0xffffffffu);
        g0[3] = (unsigned)((ga >> 32) & 0x01ffffffu) | (2u << 30);
        // D# group 1: data_size=3 (8B), 1-D tensor/tile of T elements
        g1[0] = 0x00030000u;                  // wg_mask=0, data_size=3
        g1[1] = (T & 0xffffu) << 16;          // tensor_dim0[15:0]
        g1[2] = ((T >> 16) & 0xffffu) | (1u << 16);  // tensor_dim0[31:16], tensor_dim1=1
        g1[3] = (T & 0xffffu) << 16;          // tile_dim0 = T
        g1[4] = 1u;                           // tile_dim1 = 1
        g1[5] = T;                            // tensor_dim0_stride = T
        g1[6] = 0u;
        g1[7] = 0u;
        asm volatile("tensor_load_to_lds %0, %1" :: "s"(g0), "s"(g1) : "memory");
        __builtin_amdgcn_s_wait_tensorcnt(0);
    }
    __syncthreads();

    int wid = (int)((blockIdx.x * blockDim.x + threadIdx.x) >> 5);
    if (wid >= Mtiles) return;
    int lane = threadIdx.x & 31;
    int rl = lane & 15, sel = lane >> 4;
    const float* arow = A + (size_t)(wid * 16 + rl) * LDA;

    v16h afr[KT];
#pragma unroll
    for (int kt = 0; kt < KT; ++kt)
        afr[kt] = load_afrag<IN_AFFINE>(arow, kt * 32, sel, ss);

#pragma unroll
    for (int nt = 0; nt < NT; ++nt) {
        v8f c = {};
#pragma unroll
        for (int kt = 0; kt < KT; ++kt) {
            v16h b = *(const v16h*)(bsh + ((nt * KT + kt) * 32 + lane) * 16);
            c = __builtin_amdgcn_wmma_f32_16x16x32_f16(false, afr[kt], false, b,
                                                       (short)0, c, false, false);
        }
        int col = nt * 16 + rl;
        float bv = bias[col];
        float* cp = Co + (size_t)(wid * 16 + sel * 8) * NCOLS + col;
        float s = 0.f, sq = 0.f;
#pragma unroll
        for (int r = 0; r < 8; ++r) {
            float v = c[r] + bv;
            if (!DO_STATS) v = relu_f(v);
            cp[r * NCOLS] = v;              // compile-time immediate offsets
            if (DO_STATS) { s += v; sq += v * v; }
        }
        if (DO_STATS) {
            s  += __shfl_xor(s, 16, 32);    // lanes j / j+16 share a channel
            sq += __shfl_xor(sq, 16, 32);
            if (sel == 0) {
                atomicAdd(stats + col, s);
                atomicAdd(stats + NCOLS + col, sq);
            }
        }
    }
}

// ---------------------------------------------------------------------------
// Host-side orchestration
// ---------------------------------------------------------------------------
static inline int cdiv(long long a, long long b) { return (int)((a + b - 1) / b); }

extern "C" void kernel_launch(void* const* d_in, const int* in_sizes, int n_in,
                              void* d_out, int out_size, void* d_ws, size_t ws_size,
                              hipStream_t stream) {
    const float* x   = (const float*)d_in[0];
    const int*   ei  = (const int*)d_in[1];
    const int*   nord = (const int*)d_in[2];
    const int*   bord = (const int*)d_in[3];
    const float* W1[3]   = {(const float*)d_in[4],  (const float*)d_in[10], (const float*)d_in[16]};
    const float* b1[3]   = {(const float*)d_in[5],  (const float*)d_in[11], (const float*)d_in[17]};
    const float* gma[3]  = {(const float*)d_in[6],  (const float*)d_in[12], (const float*)d_in[18]};
    const float* bet[3]  = {(const float*)d_in[7],  (const float*)d_in[13], (const float*)d_in[19]};
    const float* W2[3]   = {(const float*)d_in[8],  (const float*)d_in[14], (const float*)d_in[20]};
    const float* b2[3]   = {(const float*)d_in[9],  (const float*)d_in[15], (const float*)d_in[21]};
    const float* Wl1 = (const float*)d_in[22];
    const float* bl1 = (const float*)d_in[23];
    const float* Wl2 = (const float*)d_in[24];
    const float* bl2 = (const float*)d_in[25];
    float* out = (float*)d_out;

    // workspace layout (floats)
    float* ws = (float*)d_ws;
    size_t off = 0;
    float* P0 = ws + off; off += (size_t)NN * DH;      // aggregation buffer
    float* P1 = ws + off; off += (size_t)NN * DH;      // h1 (pre-BN)
    float* P2 = ws + off; off += (size_t)NN * DH;      // layer output
    float* HG  = ws + off; off += (size_t)GG * DH;     // graph pooled
    float* HGA = ws + off; off += (size_t)GG * DL;     // graph head activations
    float* HC  = ws + off; off += (size_t)CC * DL;     // combo pooled
    float* STATS = ws + off; off += 128;
    float* SS    = ws + off; off += 128;
    _Float16* WP = (_Float16*)(ws + off);
    _Float16* W1p[3], *W2p[3];
    size_t hoff = 0;
    W1p[0] = WP + hoff; hoff += (size_t)4 * 1 * 32 * 16;   // 30(pad32) x 64, KT=1
    for (int l = 0; l < 3; ++l) {
        if (l) { W1p[l] = WP + hoff; hoff += (size_t)4 * 2 * 32 * 16; }
        W2p[l] = WP + hoff; hoff += (size_t)4 * 2 * 32 * 16;
    }
    _Float16* Wl1p = WP + hoff; hoff += (size_t)8 * 2 * 32 * 16;   // 64 x 128

    const int TB = 256;

    // ---- pack weights into WMMA B-fragment layout ----
    pack_w_kernel<<<cdiv(4 * 1 * 32 * 8, TB), TB, 0, stream>>>(W1[0], DIN, DH, 1, W1p[0]);
    for (int l = 0; l < 3; ++l) {
        if (l) pack_w_kernel<<<cdiv(4 * 2 * 32 * 8, TB), TB, 0, stream>>>(W1[l], DH, DH, 2, W1p[l]);
        pack_w_kernel<<<cdiv(4 * 2 * 32 * 8, TB), TB, 0, stream>>>(W2[l], DH, DH, 2, W2p[l]);
    }
    pack_w_kernel<<<cdiv(8 * 2 * 32 * 8, TB), TB, 0, stream>>>(Wl1, DH, DL, 2, Wl1p);

    const int Mtiles = NN / 16;        // 25000
    const int gemmBlocks = cdiv((long long)Mtiles * 32, TB);

    // ---- 3 GIN layers ----
    for (int l = 0; l < 3; ++l) {
        int D      = (l == 0) ? DIN : DH;    // logical input dim
        int inStr  = (l == 0) ? DIN : DH;    // source row stride
        int aggStr = (l == 0) ? 32  : DH;    // aggregation buffer stride (padded K)
        const float* src = (l == 0) ? x : P2;

        // aggr = x  (+ zero pad for layer 0)
        if (l == 0)
            padcopy_x_kernel<<<cdiv((long long)NN * 32, TB), TB, 0, stream>>>(x, P0, NN * 32);
        else
            copy_f32v4_kernel<<<cdiv((long long)NN * DH / 4, TB), TB, 0, stream>>>(P2, P0, NN * DH / 4);

        // aggr[dst] += x[src]
        edge_scatter_kernel<<<cdiv((long long)EE * ((D + 3) / 4), TB), TB, 0, stream>>>(
            ei, src, inStr, D, P0, aggStr);

        // h1 = aggr @ W1 + b1, fused sum/sumsq stats
        zero_f32_kernel<<<1, 128, 0, stream>>>(STATS, 128);
        if (l == 0)
            gemm_wmma_kernel<1, DH, false, true><<<gemmBlocks, TB, 0, stream>>>(
                P0, Mtiles, W1p[0], b1[0], nullptr, P1, STATS);
        else
            gemm_wmma_kernel<2, DH, false, true><<<gemmBlocks, TB, 0, stream>>>(
                P0, Mtiles, W1p[l], b1[l], nullptr, P1, STATS);

        // fold BN into scale/shift
        bn_finalize_kernel<<<1, 64, 0, stream>>>(STATS, gma[l], bet[l], SS, 1.0f / NN);

        // out = relu( relu(bn(h1)) @ W2 + b2 )
        gemm_wmma_kernel<2, DH, true, false><<<gemmBlocks, TB, 0, stream>>>(
            P1, Mtiles, W2p[l], b2[l], SS, P2, nullptr);
    }

    // ---- pool nodes -> graphs ----
    zero_f32_kernel<<<cdiv((long long)GG * DH, TB), TB, 0, stream>>>(HG, GG * DH);
    segment_pool_kernel<<<cdiv((long long)NN * (DH / 4), TB), TB, 0, stream>>>(nord, NN, P2, DH, HG);

    // hga = relu(hg @ Wl1 + bl1)
    {
        int Mt = GG / 16;  // 1250
        gemm_wmma_kernel<2, DL, false, false><<<cdiv((long long)Mt * 32, TB), TB, 0, stream>>>(
            HG, Mt, Wl1p, bl1, nullptr, HGA, nullptr);
    }

    // ---- pool graphs -> combinations ----
    zero_f32_kernel<<<cdiv((long long)CC * DL, TB), TB, 0, stream>>>(HC, CC * DL);
    segment_pool_kernel<<<cdiv((long long)GG * (DL / 4), TB), TB, 0, stream>>>(bord, GG, HGA, DL, HC);

    // ---- final: sigmoid(hc @ Wl2 + bl2) ----
    final_dot_kernel<<<cdiv((long long)CC * 32, TB), TB, 0, stream>>>(HC, Wl2, bl2, out);
}
